// equivGNN_20968030339293
// MI455X (gfx1250) — compile-verified, matrix operand order checked by reading, not run on hardware
//
#include <hip/hip_runtime.h>

typedef __attribute__((ext_vector_type(16))) __bf16 v16bf;
typedef __attribute__((ext_vector_type(8)))  float  v8f;

union FragU { unsigned short s[16]; uint4 q[2]; v16bf v; };

__device__ __forceinline__ unsigned short f2bf(float f) {
  union { float f; unsigned u; } c; c.f = f;
  unsigned u = c.u + 0x7fffu + ((c.u >> 16) & 1u);   // round-to-nearest-even
  return (unsigned short)(u >> 16);
}

// K-map for 16-bit 16x16x32 WMMA fragments (ISA 7.12.2).
__device__ __forceinline__ int amap_k(int lane, int e) {
  return ((lane >> 4) << 3) + e + ((e >= 8) ? 8 : 0);
}

__device__ __forceinline__ v16bf b_frag(const unsigned short* pack, int frag) {
  int lane = threadIdx.x & 31;
  return *(const v16bf*)(pack + (size_t)frag * 512 + lane * 16);
}

__device__ __forceinline__ v8f wmma_bf(v16bf a, v16bf b, v8f c) {
  return __builtin_amdgcn_wmma_f32_16x16x32_bf16(false, a, false, b, (short)0, c, false, false);
}

// A-fragment from row-major bf16 global memory (two 16B loads per lane).
__device__ __forceinline__ v16bf a_frag_bf16g(const unsigned short* A, int row0, int lda, int kt) {
  FragU u;
  int lane = threadIdx.x & 31;
  int kb = (lane >> 4) << 3;
  const unsigned short* p = A + (size_t)(row0 + (lane & 15)) * lda + kt * 32;
  u.q[0] = *(const uint4*)(p + kb);
  u.q[1] = *(const uint4*)(p + 16 + kb);
  return u.v;
}

// A-fragment from a bf16 LDS tile act[16][ld].
__device__ __forceinline__ v16bf a_frag_lds16(const unsigned short* act, int ld, int kt) {
  FragU u;
  int lane = threadIdx.x & 31;
  int kb = (lane >> 4) << 3;
  const unsigned short* p = act + (lane & 15) * ld + kt * 32;
  u.q[0] = *(const uint4*)(p + kb);
  u.q[1] = *(const uint4*)(p + 16 + kb);
  return u.v;
}

// A-fragment for the K=8 edge-scalar layer: only lanes 0-15, elems 0-7 nonzero.
__device__ __forceinline__ v16bf a_frag_es8(const unsigned short* es8b, int eb) {
  FragU u;
  int lane = threadIdx.x & 31;
  uint4 z = make_uint4(0u, 0u, 0u, 0u);
  if (lane < 16) u.q[0] = *(const uint4*)(es8b + (size_t)(eb + lane) * 8);
  else           u.q[0] = z;
  u.q[1] = z;
  return u.v;
}

// Store a 16x16 D tile into a bf16 LDS tile, optional SiLU.
__device__ __forceinline__ void store_tile_lds16(unsigned short* buf, int ld, int nt, v8f acc,
                                                 float scale, bool do_silu) {
  int lane = threadIdx.x & 31;
  int col = nt * 16 + (lane & 15);
  int rb = (lane >> 4) << 3;
#pragma unroll
  for (int e = 0; e < 8; ++e) {
    float v = acc[e] * scale;
    if (do_silu) v = v / (1.f + expf(-v));
    buf[(rb + e) * ld + col] = f2bf(v);
  }
}

// Store a 16x16 D tile into an f32 LDS tile (no activation).
__device__ __forceinline__ void store_tile_ldsf(float* buf, int ld, int nt, v8f acc, float scale) {
  int lane = threadIdx.x & 31;
  int col = nt * 16 + (lane & 15);
  int rb = (lane >> 4) << 3;
#pragma unroll
  for (int e = 0; e < 8; ++e) buf[(rb + e) * ld + col] = acc[e] * scale;
}

// ---------------------------------------------------------------- pack weights
// Row-major f32 W[K][Nn] -> bf16 B-fragments [kt][nt][lane][16], K zero-padded to Kpad.
__global__ void pack_b_kernel(const float* __restrict__ W, unsigned short* __restrict__ out,
                              int K, int Nn, int Kpad) {
  int total = (Kpad / 32) * (Nn / 16) * 512;
  int ntiles = Nn / 16;
  for (int idx = blockIdx.x * blockDim.x + threadIdx.x; idx < total;
       idx += gridDim.x * blockDim.x) {
    int f = idx >> 9;
    int lane = (idx >> 4) & 31;
    int e = idx & 15;
    int kt = f / ntiles, nt = f % ntiles;
    int k = kt * 32 + amap_k(lane, e);
    int n = nt * 16 + (lane & 15);
    float v = (k < K) ? W[(long)k * Nn + n] : 0.f;
    out[idx] = f2bf(v);
  }
}

// ---------------------------------------------------------------- small utility kernels
__global__ void zero_kernel(float* __restrict__ p, long n) {
  long i = (long)blockIdx.x * blockDim.x + threadIdx.x;
  if (i < n) p[i] = 0.f;
}

__global__ void cvt_bf16_kernel(const float* __restrict__ src, unsigned short* __restrict__ dst,
                                long n) {
  long i = (long)blockIdx.x * blockDim.x + threadIdx.x;
  if (i < n) dst[i] = f2bf(src[i]);
}

// (N,128,3) f32 -> 3 planes of (N,128) bf16
__global__ void cvt_nv_planes_kernel(const float* __restrict__ nv, unsigned short* __restrict__ dst,
                                     int N) {
  long idx = (long)blockIdx.x * blockDim.x + threadIdx.x;
  if (idx >= (long)N * 384) return;
  int n = (int)(idx / 384);
  int rem = (int)(idx % 384);
  int u = rem / 3, i = rem % 3;
  dst[((long)i * N + n) * 128 + u] = f2bf(nv[idx]);
}

// ---------------------------------------------------------------- edge basis
__global__ void edge_basis_kernel(const float* __restrict__ evec, unsigned short* __restrict__ es8b,
                                  float* __restrict__ shv, int E) {
  int e = blockIdx.x * blockDim.x + threadIdx.x;
  if (e >= E) return;
  float vx = evec[(long)e * 3], vy = evec[(long)e * 3 + 1], vz = evec[(long)e * 3 + 2];
  float d = sqrtf(vx * vx + vy * vy + vz * vz);
  const float c = 4.0f;              // MAX_R - MIN_R
  float xs = d - 0.25f;              // MIN_R
  float safe = (xs > 0.f) ? xs : 1.f;
  float m = (xs > 0.f && xs < c) ? 1.f : 0.f;
  float pref = sqrtf(2.f / c) * 2.8284271247461903f * m / safe;  // sqrt(2/c)*sqrt(8)*mask/safe
  const float pi = 3.14159265358979323846f;
#pragma unroll
  for (int k = 0; k < 8; ++k)
    es8b[(long)e * 8 + k] = f2bf(pref * sinf((k + 1) * pi * safe / c));
  float dn = (d > 0.f) ? d : 1.f;
  const float sq3 = 1.7320508075688772f;
  shv[(long)e * 3 + 0] = sq3 * vx / dn;
  shv[(long)e * 3 + 1] = sq3 * vy / dn;
  shv[(long)e * 3 + 2] = sq3 * vz / dn;
}

// ---------------------------------------------------------------- generic WMMA GEMM
// A: row-major bf16 (lda elems); C: strided f32 (+ optional addend, optional bf16 copy).
// One wave per 16-row tile; NKT = K/32 compile-time. Even/odd n-tiles use two static
// fragment buffers so next-tile loads overlap WMMAs without register rotation.
template <int NKT>
__global__ __launch_bounds__(128, 1) void gemm_bf16_kernel(
    const unsigned short* __restrict__ A, int lda,
    const unsigned short* __restrict__ Bp,
    float* __restrict__ C, long csr, long csc,
    const float* __restrict__ addend, long adr, long adc,
    unsigned short* __restrict__ Cb, int ldcb,
    int M, int Nn, float scale) {
  int wave = blockIdx.x * (blockDim.x >> 5) + (threadIdx.x >> 5);
  int rowTiles = (M + 15) >> 4;
  if (wave >= rowTiles) return;
  int row0 = wave * 16;
  int lane = threadIdx.x & 31;
  v16bf afr[NKT];
#pragma unroll
  for (int kt = 0; kt < NKT; ++kt) afr[kt] = a_frag_bf16g(A, row0, lda, kt);
  int ntiles = Nn >> 4;  // even (8 or 4) at every call site

  auto store_d = [&](int nt, v8f acc) {
    long col = nt * 16 + (lane & 15);
    int rb = row0 + ((lane >> 4) << 3);
#pragma unroll
    for (int e = 0; e < 8; ++e) {
      long r = rb + e;
      float v = acc[e] * scale;
      if (addend) v += addend[r * adr + col * adc];
      C[r * csr + col * csc] = v;
      if (Cb) Cb[r * (long)ldcb + col] = f2bf(v);
    }
  };

  v16bf bA[NKT], bB[NKT];
#pragma unroll
  for (int kt = 0; kt < NKT; ++kt) bA[kt] = b_frag(Bp, kt * ntiles);
  for (int nt = 0; nt < ntiles; nt += 2) {
#pragma unroll
    for (int kt = 0; kt < NKT; ++kt) bB[kt] = b_frag(Bp, kt * ntiles + nt + 1);
    {
      v8f acc = {};
#pragma unroll
      for (int kt = 0; kt < NKT; ++kt) acc = wmma_bf(afr[kt], bA[kt], acc);
      store_d(nt, acc);
    }
    if (nt + 2 < ntiles) {
#pragma unroll
      for (int kt = 0; kt < NKT; ++kt) bA[kt] = b_frag(Bp, kt * ntiles + nt + 2);
    }
    {
      v8f acc = {};
#pragma unroll
      for (int kt = 0; kt < NKT; ++kt) acc = wmma_bf(afr[kt], bB[kt], acc);
      store_d(nt + 1, acc);
    }
  }
}

// ---------------------------------------------------------------- sc contraction
// out[n,w] = scale * sum_{u,v} feat[n,u]*attr[n,v]*SC[u,v,w];  SC packed as (F*16)x128.
// One wave per 16-node tile. Per K-step: issue all 8 B-fragment loads, build the
// rank-1 A-fragment (VALU overlaps loads), then 8 WMMAs. ~150 VGPRs, no spills.
__global__ __launch_bounds__(32, 1) void sc_kernel(const float* __restrict__ feat, int F,
                                                   const float* __restrict__ attr,
                                                   const unsigned short* __restrict__ Bp,
                                                   float* __restrict__ out, float scale) {
  __shared__ float featL[16 * 128];
  int lane = threadIdx.x;
  int nb = blockIdx.x * 16;
  for (int idx = lane; idx < 16 * F; idx += 32)
    featL[idx] = feat[(long)(nb + idx / F) * F + (idx % F)];
  __syncthreads();
  int r = lane & 15;
  int kb = (lane >> 4) << 3;
  float areg[8];
#pragma unroll
  for (int v = 0; v < 8; ++v) areg[v] = attr[(long)(nb + r) * 16 + kb + v];

  v8f acc[8] = {};
  int nkt = (F * 16) / 32;  // 64 or 32
  for (int kt = 0; kt < nkt; ++kt) {
    v16bf bf[8];
#pragma unroll
    for (int nt = 0; nt < 8; ++nt) bf[nt] = b_frag(Bp, kt * 8 + nt);
    float2 f2 = *(const float2*)&featL[r * F + 2 * kt];
    FragU u;
#pragma unroll
    for (int e = 0; e < 8; ++e) u.s[e] = f2bf(f2.x * areg[e]);
#pragma unroll
    for (int e = 0; e < 8; ++e) u.s[8 + e] = f2bf(f2.y * areg[e]);
#pragma unroll
    for (int nt = 0; nt < 8; ++nt) acc[nt] = wmma_bf(u.v, bf[nt], acc[nt]);
  }
  int rb = (lane >> 4) << 3;
#pragma unroll
  for (int nt = 0; nt < 8; ++nt) {
    long col = nt * 16 + (lane & 15);
#pragma unroll
    for (int e = 0; e < 8; ++e)
      out[(long)(nb + rb + e) * 128 + col] = acc[nt][e] * scale;
  }
}

// ---------------------------------------------------------------- fused conv1 edges
// 256-thread block = 8 waves, 8 edge-tiles. Packed MLP weights staged once into
// dynamic LDS and shared by all waves; per-wave bf16 activation tiles + f32 w-buffer.
// LDS: 45056 B weights + 8*4096 B act + 8*16384 B wbuf = 208896 B (<320KB WGP LDS).
__global__ void conv1_edge_kernel(const unsigned short* __restrict__ es8b,
                                  const float* __restrict__ shv,
                                  const int* __restrict__ esrc, const int* __restrict__ edst,
                                  const float* __restrict__ h,
                                  const unsigned short* __restrict__ pw0,
                                  const unsigned short* __restrict__ pw1,
                                  const unsigned short* __restrict__ pw2,
                                  float* __restrict__ n_s, float* __restrict__ n_v) {
  extern __shared__ char smem[];
  unsigned short* w0L = (unsigned short*)smem;       // 2048 el
  unsigned short* w1L = w0L + 2048;                  // 4096 el
  unsigned short* w2L = w1L + 4096;                  // 16384 el
  unsigned short* actB = w2L + 16384;
  int wv = threadIdx.x >> 5;
  int lane = threadIdx.x & 31;
  unsigned short* act1 = actB + wv * 2048;
  unsigned short* act2 = act1 + 1024;
  float* wbuf = (float*)(actB + 8 * 2048) + wv * (16 * 256);

  for (int i = threadIdx.x; i < 2048 / 8; i += 256) ((uint4*)w0L)[i] = ((const uint4*)pw0)[i];
  for (int i = threadIdx.x; i < 4096 / 8; i += 256) ((uint4*)w1L)[i] = ((const uint4*)pw1)[i];
  for (int i = threadIdx.x; i < 16384 / 8; i += 256) ((uint4*)w2L)[i] = ((const uint4*)pw2)[i];
  __syncthreads();

  int eb = (blockIdx.x * 8 + wv) * 16;
  const float s0 = 0.35355339059327373f;  // 8^-0.5
  const float s1 = 0.125f;                // 64^-0.5

  v16bf a0 = a_frag_es8(es8b, eb);
#pragma unroll
  for (int nt = 0; nt < 4; ++nt) {
    v8f acc = {};
    acc = wmma_bf(a0, b_frag(w0L, nt), acc);
    store_tile_lds16(act1, 64, nt, acc, s0, true);
  }
  __syncthreads();
  v16bf a1k0 = a_frag_lds16(act1, 64, 0), a1k1 = a_frag_lds16(act1, 64, 1);
#pragma unroll
  for (int nt = 0; nt < 4; ++nt) {
    v8f acc = {};
    acc = wmma_bf(a1k0, b_frag(w1L, nt), acc);
    acc = wmma_bf(a1k1, b_frag(w1L, 4 + nt), acc);
    store_tile_lds16(act2, 64, nt, acc, s1, true);
  }
  __syncthreads();
  v16bf a2k0 = a_frag_lds16(act2, 64, 0), a2k1 = a_frag_lds16(act2, 64, 1);
#pragma unroll
  for (int nt = 0; nt < 16; ++nt) {
    v8f acc = {};
    acc = wmma_bf(a2k0, b_frag(w2L, nt), acc);
    acc = wmma_bf(a2k1, b_frag(w2L, 16 + nt), acc);
    store_tile_ldsf(wbuf, 256, nt, acc, s1);
  }
  __syncthreads();
  const float inv = 0.31622776601683794f;  // 10^-0.5
  for (int idx = lane; idx < 16 * 128; idx += 32) {
    int el = idx >> 7, c = idx & 127;
    int e = eb + el;
    int src = esrc[e], dst = edst[e];
    float w0v = wbuf[el * 256 + c];
    float w1v = wbuf[el * 256 + 128 + c];
    float hv = h[(long)src * 128 + c];
    atomicAdd(&n_s[(long)dst * 128 + c], w0v * hv * inv);
    float t = w1v * hv * inv;
    float sx = shv[(long)e * 3 + 0], sy = shv[(long)e * 3 + 1], sz = shv[(long)e * 3 + 2];
    atomicAdd(&n_v[(long)dst * 384 + c * 3 + 0], t * sx);
    atomicAdd(&n_v[(long)dst * 384 + c * 3 + 1], t * sy);
    atomicAdd(&n_v[(long)dst * 384 + c * 3 + 2], t * sz);
  }
}

// ---------------------------------------------------------------- fused conv2 edges
// LDS: 28672 B weights + 8*4096 B act + 8*8192 B wbuf = 126976 B.
__global__ void conv2_edge_kernel(const unsigned short* __restrict__ es8b,
                                  const float* __restrict__ shv,
                                  const int* __restrict__ esrc, const int* __restrict__ edst,
                                  const float* __restrict__ h_s, const float* __restrict__ h_v,
                                  const unsigned short* __restrict__ pw0,
                                  const unsigned short* __restrict__ pw1,
                                  const unsigned short* __restrict__ pw2,
                                  float* __restrict__ n_mid) {
  extern __shared__ char smem[];
  unsigned short* w0L = (unsigned short*)smem;       // 2048 el
  unsigned short* w1L = w0L + 2048;                  // 4096 el
  unsigned short* w2L = w1L + 4096;                  // 8192 el
  unsigned short* actB = w2L + 8192;
  int wv = threadIdx.x >> 5;
  int lane = threadIdx.x & 31;
  unsigned short* act1 = actB + wv * 2048;
  unsigned short* act2 = act1 + 1024;
  float* wbuf = (float*)(actB + 8 * 2048) + wv * (16 * 128);

  for (int i = threadIdx.x; i < 2048 / 8; i += 256) ((uint4*)w0L)[i] = ((const uint4*)pw0)[i];
  for (int i = threadIdx.x; i < 4096 / 8; i += 256) ((uint4*)w1L)[i] = ((const uint4*)pw1)[i];
  for (int i = threadIdx.x; i < 8192 / 8; i += 256) ((uint4*)w2L)[i] = ((const uint4*)pw2)[i];
  __syncthreads();

  int eb = (blockIdx.x * 8 + wv) * 16;
  const float s0 = 0.35355339059327373f;
  const float s1 = 0.125f;

  v16bf a0 = a_frag_es8(es8b, eb);
#pragma unroll
  for (int nt = 0; nt < 4; ++nt) {
    v8f acc = {};
    acc = wmma_bf(a0, b_frag(w0L, nt), acc);
    store_tile_lds16(act1, 64, nt, acc, s0, true);
  }
  __syncthreads();
  v16bf a1k0 = a_frag_lds16(act1, 64, 0), a1k1 = a_frag_lds16(act1, 64, 1);
#pragma unroll
  for (int nt = 0; nt < 4; ++nt) {
    v8f acc = {};
    acc = wmma_bf(a1k0, b_frag(w1L, nt), acc);
    acc = wmma_bf(a1k1, b_frag(w1L, 4 + nt), acc);
    store_tile_lds16(act2, 64, nt, acc, s1, true);
  }
  __syncthreads();
  v16bf a2k0 = a_frag_lds16(act2, 64, 0), a2k1 = a_frag_lds16(act2, 64, 1);
#pragma unroll
  for (int nt = 0; nt < 8; ++nt) {
    v8f acc = {};
    acc = wmma_bf(a2k0, b_frag(w2L, nt), acc);
    acc = wmma_bf(a2k1, b_frag(w2L, 8 + nt), acc);
    store_tile_ldsf(wbuf, 128, nt, acc, s1);
  }
  __syncthreads();
  const float inv = 0.31622776601683794f;
  const float isq3 = 0.5773502691896258f;
  for (int idx = lane; idx < 16 * 64; idx += 32) {
    int el = idx >> 6, c = idx & 63;
    int e = eb + el;
    int src = esrc[e], dst = edst[e];
    float w0v = wbuf[el * 128 + c];
    float w1v = wbuf[el * 128 + 64 + c];
    float sx = shv[(long)e * 3 + 0], sy = shv[(long)e * 3 + 1], sz = shv[(long)e * 3 + 2];
    float esv = w0v * h_s[(long)src * 64 + c] * inv;
    float dot = h_v[(long)src * 192 + c * 3 + 0] * sx +
                h_v[(long)src * 192 + c * 3 + 1] * sy +
                h_v[(long)src * 192 + c * 3 + 2] * sz;
    float edv = w1v * dot * isq3 * inv;
    atomicAdd(&n_mid[(long)dst * 128 + c], esv);
    atomicAdd(&n_mid[(long)dst * 128 + 64 + c], edv);
  }
}

// ---------------------------------------------------------------- gate nonlinearity
__global__ void gate_kernel(const float* __restrict__ out_s, const float* __restrict__ out_v,
                            float* __restrict__ g_s, unsigned short* __restrict__ gsb,
                            unsigned short* __restrict__ gvb, int N) {
  int idx = blockIdx.x * blockDim.x + threadIdx.x;
  if (idx >= N * 64) return;
  int n = idx >> 6, c = idx & 63;
  float a = out_s[(long)n * 128 + c];
  float sl = a / (1.f + expf(-a));
  g_s[idx] = sl;
  gsb[idx] = f2bf(sl);
  float b = out_s[(long)n * 128 + 64 + c];
  float sig = 1.f / (1.f + expf(-b));
#pragma unroll
  for (int i = 0; i < 3; ++i)
    gvb[((long)i * N + n) * 64 + c] = f2bf(sig * out_v[(long)n * 192 + c * 3 + i]);
}

// ---------------------------------------------------------------- pooling + readout
__global__ void pool_kernel(const float* __restrict__ nodes, const int* __restrict__ batch,
                            float* __restrict__ pooled, float* __restrict__ cnt, int N) {
  int idx = blockIdx.x * blockDim.x + threadIdx.x;
  if (idx >= N * 128) return;
  int n = idx >> 7, c = idx & 127;
  int g = batch[n];
  atomicAdd(&pooled[g * 128 + c], nodes[idx]);
  if (c == 0) atomicAdd(&cnt[g], 1.f);
}

__global__ void readout_kernel(const float* __restrict__ pooled, const float* __restrict__ cnt,
                               const float* __restrict__ Wr1, const float* __restrict__ Wr2,
                               float* __restrict__ out) {
  __shared__ float pl[16 * 128];
  __shared__ float mid[16 * 128];
  int t = threadIdx.x;  // 128 threads
  for (int idx = t; idx < 2048; idx += 128) {
    int g = idx >> 7;
    float cc = cnt[g];
    if (cc < 1.f) cc = 1.f;
    pl[idx] = pooled[idx] / cc;
  }
  __syncthreads();
  const float inv = 0.08838834764831845f;  // 128^-0.5
  for (int g = 0; g < 16; ++g) {
    float acc = 0.f;
    for (int k = 0; k < 128; ++k) acc += pl[g * 128 + k] * Wr1[k * 128 + t];
    float v = acc * inv;
    mid[g * 128 + t] = v / (1.f + expf(-v));
  }
  __syncthreads();
  if (t < 16) {
    float acc = 0.f;
    for (int j = 0; j < 128; ++j) acc += mid[t * 128 + j] * Wr2[j];
    out[t] = acc * inv;
  }
}

// ================================================================ host
extern "C" void kernel_launch(void* const* d_in, const int* in_sizes, int n_in,
                              void* d_out, int out_size, void* d_ws, size_t ws_size,
                              hipStream_t stream) {
  (void)n_in; (void)out_size; (void)ws_size;
  const float* x         = (const float*)d_in[0];
  const float* node_attr = (const float*)d_in[1];
  const int*   esrc      = (const int*)d_in[2];
  const int*   edst      = (const int*)d_in[3];
  const float* evec      = (const float*)d_in[4];
  const int*   batch     = (const int*)d_in[5];
  const float* W_embed   = (const float*)d_in[6];
  const float* c1_lin1   = (const float*)d_in[7];
  const float* c1_fc_w0  = (const float*)d_in[8];
  const float* c1_fc_w1  = (const float*)d_in[9];
  const float* c1_fc_w2  = (const float*)d_in[10];
  const float* c1_lin2_s = (const float*)d_in[11];
  const float* c1_lin2_v = (const float*)d_in[12];
  const float* c1_sc     = (const float*)d_in[13];
  const float* c2_lin1_s = (const float*)d_in[14];
  const float* c2_lin1_v = (const float*)d_in[15];
  const float* c2_fc_w0  = (const float*)d_in[16];
  const float* c2_fc_w1  = (const float*)d_in[17];
  const float* c2_fc_w2  = (const float*)d_in[18];
  const float* c2_lin2   = (const float*)d_in[19];
  const float* c2_sc     = (const float*)d_in[20];
  const float* Wr1       = (const float*)d_in[21];
  const float* Wr2       = (const float*)d_in[22];
  float* out = (float*)d_out;

  const int N = in_sizes[0] / 32;   // 10000
  const int E = in_sizes[2];        // 160000

  char* ws = (char*)d_ws;
  size_t off = 0;
  auto alloc = [&](size_t bytes) -> size_t {
    off = (off + 255) & ~(size_t)255;
    size_t o = off;
    off += bytes;
    return o;
  };
  // f32 buffers
  float* shv    = (float*)(ws + alloc((size_t)E * 3 * 4));
  float* s_buf  = (float*)(ws + alloc((size_t)N * 128 * 4));
  float* h_buf  = (float*)(ws + alloc((size_t)N * 128 * 4));
  float* sc_buf = (float*)(ws + alloc((size_t)N * 128 * 4));
  float* out_s  = (float*)(ws + alloc((size_t)N * 128 * 4));
  float* out_v  = (float*)(ws + alloc((size_t)N * 192 * 4));
  float* g_s    = (float*)(ws + alloc((size_t)N * 64 * 4));
  float* h_s    = (float*)(ws + alloc((size_t)N * 64 * 4));
  float* h_v    = (float*)(ws + alloc((size_t)N * 192 * 4));
  // atomic-accumulated region (zeroed every call), contiguous:
  size_t zoff = (off + 255) & ~(size_t)255;
  float* n_s    = (float*)(ws + alloc((size_t)N * 128 * 4));
  float* n_v    = (float*)(ws + alloc((size_t)N * 384 * 4));
  float* n_mid  = (float*)(ws + alloc((size_t)N * 128 * 4));
  float* pooled = (float*)(ws + alloc(2048 * 4));
  float* cnt    = (float*)(ws + alloc(16 * 4));
  long zero_floats = (long)((off - zoff) / 4);
  // bf16 operand shadows
  unsigned short* es8b  = (unsigned short*)(ws + alloc((size_t)E * 8 * 2));
  unsigned short* xb    = (unsigned short*)(ws + alloc((size_t)N * 32 * 2));
  unsigned short* sb    = (unsigned short*)(ws + alloc((size_t)N * 128 * 2));
  unsigned short* nsb   = (unsigned short*)(ws + alloc((size_t)N * 128 * 2));
  unsigned short* nvb   = (unsigned short*)(ws + alloc((size_t)N * 384 * 2));
  unsigned short* gsb   = (unsigned short*)(ws + alloc((size_t)N * 64 * 2));
  unsigned short* gvb   = (unsigned short*)(ws + alloc((size_t)N * 192 * 2));
  unsigned short* nmidb = (unsigned short*)(ws + alloc((size_t)N * 128 * 2));
  // packed bf16 weight fragments
  unsigned short* pEmb   = (unsigned short*)(ws + alloc(4096 * 2));
  unsigned short* pL1    = (unsigned short*)(ws + alloc(16384 * 2));
  unsigned short* p1w0   = (unsigned short*)(ws + alloc(2048 * 2));
  unsigned short* p1w1   = (unsigned short*)(ws + alloc(4096 * 2));
  unsigned short* p1w2   = (unsigned short*)(ws + alloc(16384 * 2));
  unsigned short* p1l2s  = (unsigned short*)(ws + alloc(16384 * 2));
  unsigned short* p1l2v  = (unsigned short*)(ws + alloc(8192 * 2));
  unsigned short* p1sc   = (unsigned short*)(ws + alloc(262144 * 2));
  unsigned short* p2l1s  = (unsigned short*)(ws + alloc(4096 * 2));
  unsigned short* p2l1v  = (unsigned short*)(ws + alloc(4096 * 2));
  unsigned short* p2w0   = (unsigned short*)(ws + alloc(2048 * 2));
  unsigned short* p2w1   = (unsigned short*)(ws + alloc(4096 * 2));
  unsigned short* p2w2   = (unsigned short*)(ws + alloc(8192 * 2));
  unsigned short* p2lin2 = (unsigned short*)(ws + alloc(16384 * 2));
  unsigned short* p2sc   = (unsigned short*)(ws + alloc(131072 * 2));

  auto pack = [&](const float* W, unsigned short* o, int K, int Nn, int Kpad) {
    int total = (Kpad / 32) * (Nn / 16) * 512;
    pack_b_kernel<<<(total + 255) / 256, 256, 0, stream>>>(W, o, K, Nn, Kpad);
  };
  auto gemm = [&](const unsigned short* A, int lda, const unsigned short* Bp,
                  float* C, long csr, long csc, const float* add, long adr, long adc,
                  unsigned short* Cb, int ldcb, int M, int K, int Nn, float scale) {
    int rowTiles = (M + 15) / 16;
    dim3 grid((rowTiles + 3) / 4);
    if (K == 32)
      gemm_bf16_kernel<1><<<grid, 128, 0, stream>>>(A, lda, Bp, C, csr, csc, add, adr, adc,
                                                    Cb, ldcb, M, Nn, scale);
    else if (K == 64)
      gemm_bf16_kernel<2><<<grid, 128, 0, stream>>>(A, lda, Bp, C, csr, csc, add, adr, adc,
                                                    Cb, ldcb, M, Nn, scale);
    else
      gemm_bf16_kernel<4><<<grid, 128, 0, stream>>>(A, lda, Bp, C, csr, csc, add, adr, adc,
                                                    Cb, ldcb, M, Nn, scale);
  };
  auto cvt = [&](const float* src, unsigned short* dst, long n) {
    cvt_bf16_kernel<<<(int)((n + 255) / 256), 256, 0, stream>>>(src, dst, n);
  };

  const float sEmb = 0.17677669529663687f;   // 32^-0.5
  const float s128 = 0.08838834764831845f;   // 128^-0.5
  const float s64  = 0.125f;                 // 64^-0.5
  const float sSc1 = 0.022097086912079608f;  // 2048^-0.5
  const float sSc2 = 0.03125f;               // 1024^-0.5

  // 0) zero atomic targets
  zero_kernel<<<(int)((zero_floats + 255) / 256), 256, 0, stream>>>(
      (float*)(ws + zoff), zero_floats);

  // 1) pack all weights to bf16 B-fragments
  pack(W_embed,   pEmb,   32,  128, 32);
  pack(c1_lin1,   pL1,    128, 128, 128);
  pack(c1_fc_w0,  p1w0,   8,   64,  32);
  pack(c1_fc_w1,  p1w1,   64,  64,  64);
  pack(c1_fc_w2,  p1w2,   64,  256, 64);
  pack(c1_lin2_s, p1l2s,  128, 128, 128);
  pack(c1_lin2_v, p1l2v,  128, 64,  128);
  pack(c1_sc,     p1sc,   2048,128, 2048);
  pack(c2_lin1_s, p2l1s,  64,  64,  64);
  pack(c2_lin1_v, p2l1v,  64,  64,  64);
  pack(c2_fc_w0,  p2w0,   8,   64,  32);
  pack(c2_fc_w1,  p2w1,   64,  64,  64);
  pack(c2_fc_w2,  p2w2,   64,  128, 64);
  pack(c2_lin2,   p2lin2, 128, 128, 128);
  pack(c2_sc,     p2sc,   1024,128, 1024);

  // 2) edge basis (bf16 scalars + f32 unit vectors)
  edge_basis_kernel<<<(E + 255) / 256, 256, 0, stream>>>(evec, es8b, shv, E);

  // 3) node embedding: s = x@W_embed (f32 + bf16 shadow); h = s@c1_lin1
  cvt(x, xb, (long)N * 32);
  gemm(xb, 32, pEmb, s_buf, 128, 1, nullptr, 0, 0, sb, 128, N, 32, 128, sEmb);
  gemm(sb, 128, pL1, h_buf, 128, 1, nullptr, 0, 0, nullptr, 0, N, 128, 128, s128);

  // 4) conv1 fused edge MLP + scatter (8 waves/block share LDS-staged weights)
  conv1_edge_kernel<<<E / 128, 256, 208896, stream>>>(es8b, shv, esrc, edst, h_buf,
                                                      p1w0, p1w1, p1w2, n_s, n_v);

  // 5) sc1; out_s = n_s@c1_lin2_s + sc1; out_v = n_v@c1_lin2_v per component
  sc_kernel<<<N / 16, 32, 0, stream>>>(s_buf, 128, node_attr, p1sc, sc_buf, sSc1);
  cvt(n_s, nsb, (long)N * 128);
  cvt_nv_planes_kernel<<<(int)(((long)N * 384 + 255) / 256), 256, 0, stream>>>(n_v, nvb, N);
  gemm(nsb, 128, p1l2s, out_s, 128, 1, sc_buf, 128, 1, nullptr, 0, N, 128, 128, s128);
  for (int i = 0; i < 3; ++i)
    gemm(nvb + (size_t)i * N * 128, 128, p1l2v, out_v + i, 192, 3, nullptr, 0, 0,
         nullptr, 0, N, 128, 64, s128);

  // 6) gate (f32 g_s for sc2, bf16 shadows for GEMM A operands)
  gate_kernel<<<(N * 64 + 255) / 256, 256, 0, stream>>>(out_s, out_v, g_s, gsb, gvb, N);

  // 7) h_s = g_s@c2_lin1_s ; h_v = g_v@c2_lin1_v per component
  gemm(gsb, 64, p2l1s, h_s, 64, 1, nullptr, 0, 0, nullptr, 0, N, 64, 64, s64);
  for (int i = 0; i < 3; ++i)
    gemm(gvb + (size_t)i * N * 64, 64, p2l1v, h_v + i, 192, 3, nullptr, 0, 0,
         nullptr, 0, N, 64, 64, s64);

  // 8) conv2 fused edge MLP + scatter
  conv2_edge_kernel<<<E / 128, 256, 126976, stream>>>(es8b, shv, esrc, edst, h_s, h_v,
                                                      p2w0, p2w1, p2w2, n_mid);

  // 9) sc2; out = n_mid@c2_lin2 + sc2 (reuse out_s for final node features)
  sc_kernel<<<N / 16, 32, 0, stream>>>(g_s, 64, node_attr, p2sc, sc_buf, sSc2);
  cvt(n_mid, nmidb, (long)N * 128);
  gemm(nmidb, 128, p2lin2, out_s, 128, 1, sc_buf, 128, 1, nullptr, 0, N, 128, 128, s128);

  // 10) pool + readout
  pool_kernel<<<(N * 128 + 255) / 256, 256, 0, stream>>>(out_s, batch, pooled, cnt, N);
  readout_kernel<<<1, 128, 0, stream>>>(pooled, cnt, Wr1, Wr2, out);
}
